// PositionEmbeddingLearned_48842368090212
// MI455X (gfx1250) — matrix operand order repaired and checked by reading, not compiled, not run on hardware
//
#include <hip/hip_runtime.h>
#include <hip/hip_bf16.h>

typedef _Float16 half16 __attribute__((ext_vector_type(16)));
typedef float    float8 __attribute__((ext_vector_type(8)));

#define T_FRAMES 32
#define HID      128
#define HOUT     256
#define BATCH    64
#define NMAXC    64
// out layout: (NMAX, B*T, H) -> local*524288 + (b*32+t)*256 + h

// ---------------- prep 1: ragged box -> (batch, local) map ----------------
__global__ void prep_map(const int* __restrict__ nper,
                         int* __restrict__ bof, int* __restrict__ locl, int N) {
    __shared__ int starts[BATCH];
    int tid = threadIdx.x;
    if (tid == 0) {
        int s = 0;
        for (int b = 0; b < BATCH; ++b) { starts[b] = s; s += nper[b]; }
    }
    __syncthreads();
    for (int i = tid; i < N; i += blockDim.x) {
        int b = 0;
        while (b < BATCH - 1 && starts[b + 1] <= i) ++b;
        bof[i]  = b;
        locl[i] = i - starts[b];
    }
}

// ---------------- prep 2: weight staging -----------------------------------
// w1t: W1 transposed to (128 x float4) so column k is one contiguous float4
// w2h: f16 B-fragments for v_wmma_f32_16x16x32_f16:
//   frag f = nt*4+kk covers B rows K=kk*32..+31, cols nt*16..+15
//   lane n holds col (n&15); h=(n<16?0:8): K = h+j for j<8, 16+h+(j-8) for j>=8 (j = 0..15 halves, 2 per VGPR)
__global__ void prep_weights(const float* __restrict__ W1,
                             const float* __restrict__ b1,
                             const float* __restrict__ b2,
                             const float* __restrict__ W2,
                             float* __restrict__ w1t,
                             float* __restrict__ b1o,
                             float* __restrict__ b2o,
                             _Float16* __restrict__ w2h) {
    int id = blockIdx.x * blockDim.x + threadIdx.x;
    const int TOTAL = 512 + 128 + 256 + 64 * 512;
    if (id >= TOTAL) return;
    if (id < 512) {                       // W1 (4,128) -> W1T flat [k*4 + c]
        int k = id >> 2, c = id & 3;
        w1t[id] = W1[c * HID + k];
    } else if (id < 640) {
        b1o[id - 512] = b1[id - 512];
    } else if (id < 896) {
        b2o[id - 640] = b2[id - 640];
    } else {
        int t    = id - 896;              // 0 .. 32767
        int f    = t >> 9;                // fragment 0..63
        int lane = (t >> 4) & 31;
        int j    = t & 15;
        int kk   = f & 3;
        int nt   = f >> 2;
        int k    = kk * 32 + ((j < 8) ? (((lane < 16) ? 0 : 8) + j) : (16 + ((lane < 16) ? 0 : 8) + (j - 8)));
        int col  = nt * 16 + (lane & 15);
        w2h[(size_t)f * 512 + lane * 16 + j] = (_Float16)W2[k * HOUT + col];
    }
}

// ---------------- zero-fill ONLY the padded slots ---------------------------
__global__ void zero_pad(const int* __restrict__ nper, float4* __restrict__ out4,
                         long total4) {
    long id = (long)blockIdx.x * blockDim.x + threadIdx.x;
    if (id >= total4) return;
    long fi    = id << 2;                       // float index
    int  local = (int)(fi >> 19);               // / (2048*256)
    int  b     = (int)((fi >> 13) & 63);        // / (32*256) % 64
    if (local >= nper[b]) {
        float4 z = {0.f, 0.f, 0.f, 0.f};
        out4[id] = z;
    }
}

// ---------------- fused MLP + scatter, WMMA f16 -----------------------------
__global__ __launch_bounds__(128)
void mlp_scatter(const float* __restrict__ bbox,
                 const float* __restrict__ w1t,
                 const float* __restrict__ b1,
                 const float* __restrict__ b2,
                 const _Float16* __restrict__ w2h,
                 const int* __restrict__ bof,
                 const int* __restrict__ locl,
                 float* __restrict__ out) {
    __shared__ float sW1T[512];
    __shared__ float sB1[HID];
    __shared__ float sB2[HOUT];
    int tid = threadIdx.x;
    for (int t = tid; t < 512;  t += 128) sW1T[t] = w1t[t];
    for (int t = tid; t < HID;  t += 128) sB1[t]  = b1[t];
    for (int t = tid; t < HOUT; t += 128) sB2[t]  = b2[t];
    __syncthreads();

    int m   = blockIdx.x;
    int r0  = m * 16;               // 16 rows = one M tile (within one box: T=32)
    int box = r0 >> 5;
    int t0  = r0 & 31;              // 0 or 16
    int b     = bof[box];
    int local = locl[box];

    int lane = tid & 31;
    int wave = tid >> 5;
    int mrow = lane & 15;

    // per-lane bbox row (lanes L and L+16 share row L)
    float4 x = ((const float4*)bbox)[r0 + mrow];
    int koff = (lane < 16) ? 0 : 8;

    // hidden layer: each lane computes its 64 A-fragment halves (relu(x@W1+b1))
    half16 afrag[4];
    #pragma unroll
    for (int kk = 0; kk < 4; ++kk) {
        half16 af;
        #pragma unroll
        for (int v = 0; v < 8; ++v) {
            int kb = (v < 4) ? (2 * v + koff) : (16 + 2 * (v - 4) + koff);
            #pragma unroll
            for (int e = 0; e < 2; ++e) {
                int k = kk * 32 + kb + e;
                float4 w = ((const float4*)sW1T)[k];
                float hv = x.x * w.x + x.y * w.y + x.z * w.z + x.w * w.w + sB1[k];
                af[2 * v + e] = (_Float16)fmaxf(hv, 0.f);
            }
        }
        afrag[kk] = af;
    }

    size_t outbase = (size_t)local * (size_t)(BATCH * T_FRAMES) * HOUT
                   + (size_t)(b * T_FRAMES + t0) * HOUT;

    #pragma unroll
    for (int ntl = 0; ntl < 4; ++ntl) {
        int nt = wave * 4 + ntl;                 // this wave's N-tile (16 cols)
        float8 acc = {0.f, 0.f, 0.f, 0.f, 0.f, 0.f, 0.f, 0.f};
        #pragma unroll
        for (int kk = 0; kk < 4; ++kk) {
            const half16* bp =
                (const half16*)(w2h + ((size_t)(nt * 4 + kk) * 512 + lane * 16));
            half16 bf = *bp;
            acc = __builtin_amdgcn_wmma_f32_16x16x32_f16(
                false, afrag[kk], false, bf, (short)0, acc, false, false);
        }
        int col   = nt * 16 + (lane & 15);
        int rbase = (lane < 16) ? 0 : 8;
        float bias = sB2[col];
        #pragma unroll
        for (int j = 0; j < 8; ++j) {
            int mr = j + rbase;                  // row within tile (= t - t0)
            out[outbase + (size_t)mr * HOUT + col] = acc[j] + bias;
        }
    }
}

extern "C" void kernel_launch(void* const* d_in, const int* in_sizes, int n_in,
                              void* d_out, int out_size, void* d_ws, size_t ws_size,
                              hipStream_t stream) {
    (void)n_in; (void)out_size; (void)ws_size;
    const float* bbox = (const float*)d_in[0];
    const float* W1   = (const float*)d_in[1];
    const float* b1   = (const float*)d_in[2];
    const float* W2   = (const float*)d_in[3];
    const float* b2   = (const float*)d_in[4];
    const int*   nper = (const int*)d_in[5];

    int NT = in_sizes[0] / 4;          // total rows  (N*T)
    int N  = NT / T_FRAMES;            // total boxes

    char* ws = (char*)d_ws;
    int*       bof  = (int*)(ws + 0);          // N ints   (<= 16 KB)
    int*       lloc = (int*)(ws + 16384);      // N ints
    float*     w1t  = (float*)(ws + 32768);    // 512 f
    float*     b1o  = (float*)(ws + 34816);    // 128 f
    float*     b2o  = (float*)(ws + 35328);    // 256 f
    _Float16*  w2h  = (_Float16*)(ws + 36352); // 32768 halves (64 KB)

    prep_map<<<1, 256, 0, stream>>>(nper, bof, lloc, N);

    const int WTOT = 512 + 128 + 256 + 64 * 512;
    prep_weights<<<(WTOT + 255) / 256, 256, 0, stream>>>(W1, b1, b2, W2,
                                                         w1t, b1o, b2o, w2h);

    long total4 = (long)NMAXC * BATCH * T_FRAMES * HOUT / 4;  // 8,388,608
    zero_pad<<<(int)((total4 + 255) / 256), 256, 0, stream>>>(
        nper, (float4*)d_out, total4);

    mlp_scatter<<<2 * N, 128, 0, stream>>>(bbox, w1t, b1o, b2o, w2h,
                                           bof, lloc, (float*)d_out);
}
